// EfficientFlashAttention_8555574854199
// MI455X (gfx1250) — matrix operand order repaired and checked by reading, not compiled, not run on hardware
//
#include <hip/hip_runtime.h>

// ---------------------------------------------------------------------------
// MHSA block for MI455X (gfx1250, wave32, WMMA, async global->LDS staging).
// x:[B,S,D] f32; Wq/Wk/Wv/Wo:[D,D] f32 ([out,in], y = x@W.T); bo:[D] f32.
// bf16 compute (v_wmma_f32_16x16x32_bf16) with f32 accumulation.
// ---------------------------------------------------------------------------

#define DEV __device__ __forceinline__

typedef __attribute__((ext_vector_type(16))) __bf16 v16bf;
typedef __attribute__((ext_vector_type(8)))  float  v8f;

constexpr int Bsz = 2, Sq = 2048, Dm = 1024, Hn = 16, HDm = 64;
constexpr int Mtot = Bsz * Sq;   // 4096 tokens

DEV unsigned f2bf(float f) {
  unsigned u = __builtin_bit_cast(unsigned, f);
  return (u + 0x7FFFu + ((u >> 16) & 1u)) >> 16;   // RNE, low 16 bits valid
}

DEV v8f zero8() { v8f z = {0.f,0.f,0.f,0.f,0.f,0.f,0.f,0.f}; return z; }

struct FragBF { union { v16bf v; unsigned u[8]; }; };

DEV v8f wmma_bf16(const FragBF& a, const FragBF& b, v8f c) {
  return __builtin_amdgcn_wmma_f32_16x16x32_bf16(
      false, a.v, false, b.v, (short)0, c, false, false);
}

// ---- CDNA5 async global->LDS copy (ASYNCcnt) ------------------------------
// ISA 10.2: flat LDS addresses map via addr[31:0] -> LDS offset, so the low
// 32 bits of a generic pointer into __shared__ give the per-lane VDST value.
DEV void async_copy_b128(void* lds_dst, const void* gsrc) {
  unsigned dst = (unsigned)(unsigned long long)lds_dst;
  unsigned long long src = (unsigned long long)gsrc;
  asm volatile("global_load_async_to_lds_b128 %0, %1, off"
               :: "v"(dst), "v"(src) : "memory");
}
DEV void wait_async0() { asm volatile("s_wait_asynccnt 0" ::: "memory"); }
DEV void wait_ds0()    { asm volatile("s_wait_dscnt 0"    ::: "memory"); }

// A-matrix 16x32 bf16 fragment (ISA 7.12.2): lane = M (dup in hi half),
// VGPR r: r<4 -> K = 2r + 8*hi ; r>=4 -> K = 16 + 2(r-4) + 8*hi.
DEV void load_a_frag(FragBF& f, const unsigned short* p, int stride, int k_base) {
  int lane = threadIdx.x & 31;
  int m = lane & 15, hi = lane >> 4;
#pragma unroll
  for (int r = 0; r < 8; ++r) {
    int k = (r < 4) ? (2 * r + 8 * hi) : (16 + 2 * (r - 4) + 8 * hi);
    f.u[r] = *(const unsigned*)(p + (size_t)m * stride + k_base + k);
  }
}

// B-matrix 32x16 bf16 fragment: lane = N, VGPR r: r<4 -> K = 2r + 8*hi ; r>=4 -> K = 16 + 2(r-4) + 8*hi (+1 packed).
// pT is [n][k] (K-contiguous) so each K-pair is one dword load.
DEV void load_b_frag(FragBF& f, const unsigned short* pT, int stride,
                     int n_base, int k_base) {
  int lane = threadIdx.x & 31;
  int n = n_base + (lane & 15);
  int hi = lane >> 4;
#pragma unroll
  for (int r = 0; r < 8; ++r) {
    int k = (r < 4) ? (2 * r + 8 * hi) : (16 + 2 * (r - 4) + 8 * hi);
    f.u[r] = *(const unsigned*)(pT + (size_t)n * stride + k_base + k);
  }
}

// C-fragment row reductions: 16 cols live on lanes 0..15 (rows 0..7) and
// lanes 16..31 (rows 8..15) -> xor-reduce within each half (wave32).
DEV float red_max16(float v) {
#pragma unroll
  for (int m = 1; m <= 8; m <<= 1) v = fmaxf(v, __shfl_xor(v, m, 32));
  return v;
}
DEV float red_sum16(float v) {
#pragma unroll
  for (int m = 1; m <= 8; m <<= 1) v += __shfl_xor(v, m, 32);
  return v;
}

// ---------------------------------------------------------------------------
// f32 -> bf16 convert, 4 elements/thread (b128 in, b64 out)
// ---------------------------------------------------------------------------
__global__ void cvt_f32_bf16_kernel(const float4* __restrict__ x,
                                    uint2* __restrict__ y, int n4) {
  int i = blockIdx.x * 256 + threadIdx.x;
  if (i < n4) {
    float4 v = x[i];
    uint2 o;
    o.x = f2bf(v.x) | (f2bf(v.y) << 16);
    o.y = f2bf(v.z) | (f2bf(v.w) << 16);
    y[i] = o;
  }
}

// W:[N out][K in] f32 -> Wt:[K][N] bf16 (GEMM B = W^T row-major [K,N])
__global__ void transpose_w_kernel(const float* __restrict__ W,
                                   unsigned short* __restrict__ Wt,
                                   int N, int K) {
  __shared__ float t[32][33];
  int k0 = blockIdx.x * 32, n0 = blockIdx.y * 32;
  int tx = threadIdx.x, ty = threadIdx.y;   // 32 x 8
#pragma unroll
  for (int j = 0; j < 4; ++j)
    t[ty + j * 8][tx] = W[(size_t)(n0 + ty + j * 8) * K + k0 + tx];
  __syncthreads();
#pragma unroll
  for (int j = 0; j < 4; ++j)
    Wt[(size_t)(k0 + ty + j * 8) * N + n0 + tx] =
        (unsigned short)f2bf(t[tx][ty + j * 8]);
}

// ---------------------------------------------------------------------------
// bf16 WMMA GEMM: C[M,N] = A[M,K] @ B[K,N].
// Workgroup (256 thr = 8 waves) computes 256x64; each wave 32x64 (2 M-tiles
// sharing each B fragment -> 8 WMMA per K-chunk per wave).
// A tile staged via global_load_async_to_lds_b128; B staged b128 + transpose.
// mode 0: bf16 scatter to [B,H,S,HD]; mode 1: f32 + bias to outF.
// ---------------------------------------------------------------------------
__global__ __launch_bounds__(256)
void gemm_bf16_kernel(const unsigned short* __restrict__ A,
                      const unsigned short* __restrict__ Bt,
                      float* __restrict__ outF,
                      unsigned short* __restrict__ outH,
                      const float* __restrict__ bias,
                      int M, int N, int K, int mode) {
  __shared__ unsigned short As[256 * 40];   // [m][k], 80B row (16B aligned)
  __shared__ unsigned short Bs[64 * 34];    // [n][k] transposed-in-LDS
  int tid = threadIdx.x;
  int wave = tid >> 5, lane = tid & 31;
  int n0 = blockIdx.x * 64, m0 = blockIdx.y * 256;

  v8f acc[2][4];
#pragma unroll
  for (int mt = 0; mt < 2; ++mt)
#pragma unroll
    for (int nt = 0; nt < 4; ++nt) acc[mt][nt] = zero8();

  for (int kc = 0; kc < K; kc += 32) {
    __syncthreads();
    // stage A 256x32: 1024 x b128 async copies (4 per thread)
#pragma unroll
    for (int i = 0; i < 4; ++i) {
      int e = tid + i * 256;            // 0..1023
      int r = e >> 2, c = e & 3;        // row, 8-elem chunk
      async_copy_b128(As + r * 40 + c * 8,
                      A + (size_t)(m0 + r) * K + kc + c * 8);
    }
    // stage B 32x64 transposed: b128 read + 8x b16 scatter per thread
    {
      int kr = tid >> 3, c8 = tid & 7;  // 32 k-rows x 8 chunks
      union { uint4 v; unsigned short s[8]; } d;
      d.v = *(const uint4*)(Bt + (size_t)(kc + kr) * N + n0 + c8 * 8);
#pragma unroll
      for (int j = 0; j < 8; ++j) Bs[(c8 * 8 + j) * 34 + kr] = d.s[j];
    }
    wait_async0();
    __syncthreads();

    FragBF af[2];
#pragma unroll
    for (int mt = 0; mt < 2; ++mt)
      load_a_frag(af[mt], As + (wave * 32 + mt * 16) * 40, 40, 0);
#pragma unroll
    for (int nt = 0; nt < 4; ++nt) {
      FragBF bf;
      load_b_frag(bf, Bs, 34, nt * 16, 0);
#pragma unroll
      for (int mt = 0; mt < 2; ++mt)
        acc[mt][nt] = wmma_bf16(af[mt], bf, acc[mt][nt]);
    }
  }

  int hi = lane >> 4, nl = lane & 15;
#pragma unroll
  for (int mt = 0; mt < 2; ++mt) {
#pragma unroll
    for (int nt = 0; nt < 4; ++nt) {
      int col = n0 + nt * 16 + nl;
#pragma unroll
      for (int r = 0; r < 8; ++r) {
        int row = m0 + wave * 32 + mt * 16 + r + hi * 8;
        float val = acc[mt][nt][r];
        if (mode == 1) {
          outF[(size_t)row * N + col] = val + bias[col];
        } else {
          int h = col >> 6, hd = col & 63;       // HD = 64
          int b = row >> 11, s = row & 2047;     // S = 2048
          outH[(((size_t)(b * Hn + h) * Sq + s) << 6) + hd] =
              (unsigned short)f2bf(val);
        }
      }
    }
  }
}

// ---------------------------------------------------------------------------
// Flash attention: grid (S/128, B*H), 256 threads = 8 waves.
// Wave owns 16 query rows; workgroup streams 32-key K/V chunks (K via async
// b128 to LDS). Per chunk/wave: 4 WMMA scores + online softmax + 4 WMMA P@V.
// ---------------------------------------------------------------------------
__global__ __launch_bounds__(256)
void flash_attn_kernel(const unsigned short* __restrict__ Qh,
                       const unsigned short* __restrict__ Kh,
                       const unsigned short* __restrict__ Vh,
                       unsigned short* __restrict__ Oflat) {
  __shared__ unsigned short Ks[32 * 72];    // [key][hd], 144B row (16B align)
  __shared__ unsigned short Vs[64 * 34];    // [hd][key] (key contiguous)
  __shared__ unsigned short Ps[8][16 * 34]; // per-wave P scratch [m][key]

  int tid = threadIdx.x, wave = tid >> 5, lane = tid & 31;
  int hi = lane >> 4, nl = lane & 15;
  int bh = blockIdx.y, b = bh >> 4, h = bh & 15;   // H = 16
  const unsigned short* Qb = Qh + (size_t)bh * Sq * HDm;
  const unsigned short* Kb = Kh + (size_t)bh * Sq * HDm;
  const unsigned short* Vb = Vh + (size_t)bh * Sq * HDm;
  int q0 = blockIdx.x * 128 + wave * 16;

  FragBF qf[2];
  load_a_frag(qf[0], Qb + (size_t)q0 * HDm, HDm, 0);
  load_a_frag(qf[1], Qb + (size_t)q0 * HDm, HDm, 32);

  float m_i[8], l_i[8];
#pragma unroll
  for (int r = 0; r < 8; ++r) { m_i[r] = -1e30f; l_i[r] = 0.f; }
  v8f o[4];
#pragma unroll
  for (int i = 0; i < 4; ++i) o[i] = zero8();

  const float scale = 0.125f;   // 1/sqrt(HD=64)

  for (int kc = 0; kc < Sq; kc += 32) {
    // stage K chunk 32x64: 256 x b128 async (1 per thread)
    {
      int key = tid >> 3, c8 = tid & 7;
      async_copy_b128(Ks + key * 72 + c8 * 8,
                      Kb + (size_t)(kc + key) * HDm + c8 * 8);
    }
    // stage V chunk transposed: b128 read + 8x b16 scatter
    {
      int key = tid >> 3, c8 = tid & 7;
      union { uint4 v; unsigned short s[8]; } d;
      d.v = *(const uint4*)(Vb + (size_t)(kc + key) * HDm + c8 * 8);
#pragma unroll
      for (int j = 0; j < 8; ++j) Vs[(c8 * 8 + j) * 34 + key] = d.s[j];
    }
    wait_async0();
    __syncthreads();

    // scores S = Q @ K^T : B-frag lane = key(n), k = hd from Ks[key][hd]
    v8f sf[2];
#pragma unroll
    for (int nt = 0; nt < 2; ++nt) {
      sf[nt] = zero8();
      FragBF kb0; load_b_frag(kb0, Ks, 72, nt * 16, 0);
      sf[nt] = wmma_bf16(qf[0], kb0, sf[nt]);
      FragBF kb1; load_b_frag(kb1, Ks, 72, nt * 16, 32);
      sf[nt] = wmma_bf16(qf[1], kb1, sf[nt]);
    }

    // online softmax per C-fragment row
    unsigned short* Pw = &Ps[wave][0];
#pragma unroll
    for (int r = 0; r < 8; ++r) {
      float s0 = sf[0][r] * scale, s1 = sf[1][r] * scale;
      float rm = red_max16(fmaxf(s0, s1));
      float mn = fmaxf(m_i[r], rm);
      float al = __expf(m_i[r] - mn);
      float p0 = __expf(s0 - mn), p1 = __expf(s1 - mn);
      l_i[r] = l_i[r] * al + red_sum16(p0 + p1);
      m_i[r] = mn;
#pragma unroll
      for (int nt2 = 0; nt2 < 4; ++nt2) o[nt2][r] *= al;
      int row = r + hi * 8;
      Pw[row * 34 + nl]      = (unsigned short)f2bf(p0);
      Pw[row * 34 + 16 + nl] = (unsigned short)f2bf(p1);
    }
    wait_ds0();   // wave-local LDS RAW fence before re-reading P in A-layout

    FragBF pf;
    load_a_frag(pf, Pw, 34, 0);
#pragma unroll
    for (int nt2 = 0; nt2 < 4; ++nt2) {
      FragBF vf;
      load_b_frag(vf, Vs, 34, nt2 * 16, 0);
      o[nt2] = wmma_bf16(pf, vf, o[nt2]);
    }
    __syncthreads();
  }

  // normalize and scatter to flat [B*S, D] bf16
#pragma unroll
  for (int r = 0; r < 8; ++r) {
    float inv = 1.f / l_i[r];
    int sIdx = q0 + r + hi * 8;
#pragma unroll
    for (int nt2 = 0; nt2 < 4; ++nt2) {
      int col = h * HDm + nt2 * 16 + nl;
      Oflat[(size_t)(b * Sq + sIdx) * Dm + col] =
          (unsigned short)f2bf(o[nt2][r] * inv);
    }
  }
}

// ---------------------------------------------------------------------------
extern "C" void kernel_launch(void* const* d_in, const int* in_sizes, int n_in,
                              void* d_out, int out_size, void* d_ws, size_t ws_size,
                              hipStream_t stream) {
  (void)in_sizes; (void)n_in; (void)out_size; (void)ws_size;
  const float* x  = (const float*)d_in[0];
  const float* Wq = (const float*)d_in[1];
  const float* Wk = (const float*)d_in[2];
  const float* Wv = (const float*)d_in[3];
  const float* Wo = (const float*)d_in[4];
  const float* bo = (const float*)d_in[5];
  float* out = (float*)d_out;

  // workspace carve (all bf16 as ushort): 24M elems = 48 MB
  unsigned short* ws = (unsigned short*)d_ws;
  size_t off = 0;
  unsigned short* xb  = ws + off; off += (size_t)Mtot * Dm;
  unsigned short* Wqt = ws + off; off += (size_t)Dm * Dm;
  unsigned short* Wkt = ws + off; off += (size_t)Dm * Dm;
  unsigned short* Wvt = ws + off; off += (size_t)Dm * Dm;
  unsigned short* Wot = ws + off; off += (size_t)Dm * Dm;
  unsigned short* Qhd = ws + off; off += (size_t)Mtot * Dm;
  unsigned short* Khd = ws + off; off += (size_t)Mtot * Dm;
  unsigned short* Vhd = ws + off; off += (size_t)Mtot * Dm;
  unsigned short* Ofl = ws + off; off += (size_t)Mtot * Dm;

  int n4 = (Mtot * Dm) / 4;
  cvt_f32_bf16_kernel<<<(n4 + 255) / 256, 256, 0, stream>>>(
      (const float4*)x, (uint2*)xb, n4);

  dim3 tb(32, 8), tg(Dm / 32, Dm / 32);
  transpose_w_kernel<<<tg, tb, 0, stream>>>(Wq, Wqt, Dm, Dm);
  transpose_w_kernel<<<tg, tb, 0, stream>>>(Wk, Wkt, Dm, Dm);
  transpose_w_kernel<<<tg, tb, 0, stream>>>(Wv, Wvt, Dm, Dm);
  transpose_w_kernel<<<tg, tb, 0, stream>>>(Wo, Wot, Dm, Dm);

  dim3 gg(Dm / 64, Mtot / 256);   // (16, 16)
  gemm_bf16_kernel<<<gg, 256, 0, stream>>>(xb, Wqt, nullptr, Qhd, nullptr,
                                           Mtot, Dm, Dm, 0);
  gemm_bf16_kernel<<<gg, 256, 0, stream>>>(xb, Wkt, nullptr, Khd, nullptr,
                                           Mtot, Dm, Dm, 0);
  gemm_bf16_kernel<<<gg, 256, 0, stream>>>(xb, Wvt, nullptr, Vhd, nullptr,
                                           Mtot, Dm, Dm, 0);

  flash_attn_kernel<<<dim3(Sq / 128, Bsz * Hn), 256, 0, stream>>>(Qhd, Khd,
                                                                  Vhd, Ofl);

  gemm_bf16_kernel<<<gg, 256, 0, stream>>>(Ofl, Wot, out, nullptr, bo,
                                           Mtot, Dm, Dm, 1);
}
